// DGCN_77163382440035
// MI455X (gfx1250) — matrix operand order repaired and checked by reading, not compiled, not run on hardware
//
#include <hip/hip_runtime.h>

typedef __attribute__((ext_vector_type(16))) __bf16 bf16x16;
typedef __attribute__((ext_vector_type(8)))  __bf16 bf16x8;
typedef __attribute__((ext_vector_type(8)))  float  v8f;

#define NEG_BIG (-3.4e38f)

__device__ __forceinline__ float lrelu(float v) { return v >= 0.f ? v : 0.2f * v; }

// ---------------------------------------------------------------------------
// Pack W[Cin,Cout] (fp32 row-major) into bf16 B-fragment order:
//   WP[((t*ksteps + s)*32 + lane)*16 + i] = W[s*32 + koffB(lane) + (i&7) + 16*(i>>3)][t*16 + (lane&15)]
// so WMMA B operands are contiguous 32-byte vectors per lane per K-step.
// ---------------------------------------------------------------------------
__global__ void dgcn_pack_w_kernel(const float* __restrict__ W, __bf16* __restrict__ WP,
                                   int Cin, int Cout) {
  int idx = blockIdx.x * blockDim.x + threadIdx.x;
  int total = Cin * Cout;
  if (idx >= total) return;
  int i    = idx & 15;
  int lane = (idx >> 4) & 31;
  int rest = idx >> 9;              // t*ksteps + s
  int ksteps = Cin >> 5;
  int t = rest / ksteps;
  int s = rest - t * ksteps;
  int koffB = (lane & 16) ? 8 : 0;
  int kk  = s * 32 + koffB + (i & 7) + ((i & 8) ? 16 : 0);
  int col = t * 16 + (lane & 15);
  WP[idx] = (__bf16)W[(size_t)kk * Cout + col];
}

// ---------------------------------------------------------------------------
// x[B,32,N] -> XR[B*N,32] fp32  +  XRb bf16 shadow (WMMA A operand)
// ---------------------------------------------------------------------------
__global__ __launch_bounds__(1024)
void dgcn_transpose_kernel(const float* __restrict__ x, float* __restrict__ xr,
                           __bf16* __restrict__ xrb, int N) {
  __shared__ float tile[32][33];
  const int n0 = blockIdx.x * 32;
  const int b  = blockIdx.y;
  tile[threadIdx.y][threadIdx.x] = x[((size_t)b * 32 + threadIdx.y) * N + n0 + threadIdx.x];
  __syncthreads();
  float v = tile[threadIdx.x][threadIdx.y];
  size_t o = ((size_t)b * N + n0 + threadIdx.y) * 32 + threadIdx.x;
  xr[o]  = v;
  xrb[o] = (__bf16)v;
}

// ---------------------------------------------------------------------------
// H = X @ W, bf16 WMMA 16x16x32 fp32-accumulate.
// Register block per wave: 2 M-tiles x 2 N-tiles = 4 WMMA / K-step.
// B panel (<=16KB) staged in LDS once per block (8 waves share it), so the
// inner loop is: 4 x global b128 (A) + 4 x ds b128 (B) -> 4 x v_wmma.
// Per-channel sum/sumsq via LDS float atomics -> 1 global atomic per block.
// Grid: (R/256, Cout/32), 256 threads = 8 waves.
// ---------------------------------------------------------------------------
__global__ __launch_bounds__(256)
void dgcn_gemm_stats_kernel(const __bf16* __restrict__ Xb, const __bf16* __restrict__ WP,
                            float* __restrict__ H, float* __restrict__ sum,
                            float* __restrict__ sumsq, int Cin, int Cout) {
  __shared__ __bf16 Bsh[8192];      // 2 tiles * (Cin<=256)/32 ksteps * 512
  __shared__ float bsum[32];
  __shared__ float bsq[32];
  const int lane   = threadIdx.x & 31;
  const int wave   = threadIdx.x >> 5;
  const int ksteps = Cin >> 5;
  const int t0     = blockIdx.y * 2;
  if (threadIdx.x < 32) { bsum[threadIdx.x] = 0.f; bsq[threadIdx.x] = 0.f; }
  {
    // cooperative copy of this block's 2-tile B panel into LDS (16B chunks)
    const uint4* src = (const uint4*)(WP + (size_t)t0 * ksteps * 512);
    uint4* dst = (uint4*)Bsh;
    const int n16 = ksteps * 128;   // 2*ksteps*512 bf16 = ksteps*128 uint4
    for (int i = threadIdx.x; i < n16; i += 256) dst[i] = src[i];
  }
  __syncthreads();

  const int rowBase = (blockIdx.x * 8 + wave) * 32;   // 2 M-tiles per wave
  const int m       = lane & 15;
  const int halfA   = (lane & 16) ? 8 : 0;
  const __bf16* arow0 = Xb + (size_t)(rowBase + m) * Cin + halfA;
  const __bf16* arow1 = Xb + (size_t)(rowBase + 16 + m) * Cin + halfA;
  const __bf16* bsh0  = Bsh + (size_t)lane * 16;

  v8f c00 = {}, c01 = {}, c10 = {}, c11 = {};
  for (int s = 0; s < ksteps; ++s) {
    bf16x16 a0, a1;
    {
      bf16x8 lo = *(const bf16x8*)(arow0 + s * 32);
      bf16x8 hi = *(const bf16x8*)(arow0 + s * 32 + 16);
#pragma unroll
      for (int i = 0; i < 8; ++i) { a0[i] = lo[i]; a0[i + 8] = hi[i]; }
    }
    {
      bf16x8 lo = *(const bf16x8*)(arow1 + s * 32);
      bf16x8 hi = *(const bf16x8*)(arow1 + s * 32 + 16);
#pragma unroll
      for (int i = 0; i < 8; ++i) { a1[i] = lo[i]; a1[i + 8] = hi[i]; }
    }
    bf16x16 b0 = *(const bf16x16*)(bsh0 + (size_t)s * 512);
    bf16x16 b1 = *(const bf16x16*)(bsh0 + (size_t)(ksteps + s) * 512);
    c00 = __builtin_amdgcn_wmma_f32_16x16x32_bf16(false, a0, false, b0, (short)0, c00, false, false);
    c01 = __builtin_amdgcn_wmma_f32_16x16x32_bf16(false, a0, false, b1, (short)0, c01, false, false);
    c10 = __builtin_amdgcn_wmma_f32_16x16x32_bf16(false, a1, false, b0, (short)0, c10, false, false);
    c11 = __builtin_amdgcn_wmma_f32_16x16x32_bf16(false, a1, false, b1, (short)0, c11, false, false);
  }

  const int mo = (lane & 16) ? 8 : 0;   // C: M = vgpr + 8*laneHalf, N = lane&15
  const int colBase = t0 * 16;
  float s0 = 0.f, q0 = 0.f, s1 = 0.f, q1 = 0.f;
#pragma unroll
  for (int r = 0; r < 8; ++r) {
    float v00 = c00[r], v01 = c01[r], v10 = c10[r], v11 = c11[r];
    float* hp0 = H + (size_t)(rowBase + r + mo) * Cout + colBase + m;
    float* hp1 = H + (size_t)(rowBase + 16 + r + mo) * Cout + colBase + m;
    hp0[0]  = v00;  hp0[16] = v01;
    hp1[0]  = v10;  hp1[16] = v11;
    s0 += v00 + v10;  q0 += v00 * v00 + v10 * v10;
    s1 += v01 + v11;  q1 += v01 * v01 + v11 * v11;
  }
  atomicAdd(&bsum[m],      s0);
  atomicAdd(&bsq[m],       q0);
  atomicAdd(&bsum[16 + m], s1);
  atomicAdd(&bsq[16 + m],  q1);
  __syncthreads();
  if (threadIdx.x < 32) {
    atomicAdd(&sum[colBase + threadIdx.x],   bsum[threadIdx.x]);
    atomicAdd(&sumsq[colBase + threadIdx.x], bsq[threadIdx.x]);
  }
}

// ---------------------------------------------------------------------------
// Layer-0 edge branch: one WMMA tile == one node's 16 edges (centers are
// repeat(arange(N),16)). A row = xr[nbr]-xr[ctr], Cin=32 -> single K-step.
// Two N-tiles per wave. Stats pass + apply pass recompute (no 134MB tensor).
// ---------------------------------------------------------------------------
__device__ __forceinline__ void dgcn_edge_tile_mm(const float* __restrict__ XR,
                                                  const __bf16* __restrict__ WP,
                                                  const int* __restrict__ eidx,
                                                  int node, int N, int E,
                                                  int t0, int lane,
                                                  v8f& c0, v8f& c1) {
  const int m     = lane & 15;
  const int b     = node / N;
  const int j     = node - b * N;
  const int nbr   = eidx[E + j * 16 + m] + b * N;   // this tile-row's neighbor
  const int halfA = (lane & 16) ? 8 : 0;
  const float* pn = XR + (size_t)nbr  * 32 + halfA;
  const float* pc = XR + (size_t)node * 32 + halfA; // center row == node
  bf16x16 a;
#pragma unroll
  for (int i = 0; i < 8; ++i) {
    a[i]     = (__bf16)(pn[i]      - pc[i]);
    a[i + 8] = (__bf16)(pn[16 + i] - pc[16 + i]);
  }
  bf16x16 b0 = *(const bf16x16*)(WP + ((size_t)(t0    ) * 32 + lane) * 16);
  bf16x16 b1 = *(const bf16x16*)(WP + ((size_t)(t0 + 1) * 32 + lane) * 16);
  c0 = {};  c1 = {};
  c0 = __builtin_amdgcn_wmma_f32_16x16x32_bf16(false, a, false, b0, (short)0, c0, false, false);
  c1 = __builtin_amdgcn_wmma_f32_16x16x32_bf16(false, a, false, b1, (short)0, c1, false, false);
}

__global__ __launch_bounds__(256)
void dgcn_edge_stats_kernel(const float* __restrict__ XR, const __bf16* __restrict__ WP,
                            const int* __restrict__ eidx, float* __restrict__ sum,
                            float* __restrict__ sumsq, int N, int E, int Cout) {
  __shared__ float bsum[32];
  __shared__ float bsq[32];
  const int lane = threadIdx.x & 31;
  const int wave = threadIdx.x >> 5;
  if (threadIdx.x < 32) { bsum[threadIdx.x] = 0.f; bsq[threadIdx.x] = 0.f; }
  __syncthreads();
  const int node = blockIdx.x * 8 + wave;
  const int t0   = blockIdx.y * 2;
  v8f c0, c1;
  dgcn_edge_tile_mm(XR, WP, eidx, node, N, E, t0, lane, c0, c1);
  const int m = lane & 15;
  float s0 = 0.f, q0 = 0.f, s1 = 0.f, q1 = 0.f;
#pragma unroll
  for (int r = 0; r < 8; ++r) {
    float v0 = c0[r], v1 = c1[r];
    s0 += v0; q0 += v0 * v0;
    s1 += v1; q1 += v1 * v1;
  }
  atomicAdd(&bsum[m],      s0);
  atomicAdd(&bsq[m],       q0);
  atomicAdd(&bsum[16 + m], s1);
  atomicAdd(&bsq[16 + m],  q1);
  __syncthreads();
  if (threadIdx.x < 32) {
    atomicAdd(&sum[t0 * 16 + threadIdx.x],   bsum[threadIdx.x]);
    atomicAdd(&sumsq[t0 * 16 + threadIdx.x], bsq[threadIdx.x]);
  }
}

__global__ __launch_bounds__(256)
void dgcn_edge_apply_kernel(const float* __restrict__ XR, const __bf16* __restrict__ WP,
                            const int* __restrict__ eidx, const float* __restrict__ scale,
                            const float* __restrict__ shift, float* __restrict__ Xout,
                            __bf16* __restrict__ Xoutb, int N, int E, int Cout) {
  const int lane = threadIdx.x & 31;
  const int wave = threadIdx.x >> 5;
  const int node = blockIdx.x * 8 + wave;
  const int t0   = blockIdx.y * 2;
  v8f c0, c1;
  dgcn_edge_tile_mm(XR, WP, eidx, node, N, E, t0, lane, c0, c1);
  const int m = lane & 15;
  const float sc0 = scale[t0 * 16 + m],      sh0 = shift[t0 * 16 + m];
  const float sc1 = scale[t0 * 16 + 16 + m], sh1 = shift[t0 * 16 + 16 + m];
  float mx0 = NEG_BIG, mx1 = NEG_BIG;
#pragma unroll
  for (int r = 0; r < 8; ++r) {
    mx0 = fmaxf(mx0, lrelu(c0[r] * sc0 + sh0));
    mx1 = fmaxf(mx1, lrelu(c1[r] * sc1 + sh1));
  }
  mx0 = fmaxf(mx0, __shfl_xor(mx0, 16, 32));   // merge tile M-halves: segment max
  mx1 = fmaxf(mx1, __shfl_xor(mx1, 16, 32));
  if (lane < 16) {
    size_t o = (size_t)node * Cout + t0 * 16 + lane;
    float n0 = Xout[o]      + mx0;   // Xout pre-filled with lrelu(BN(h_a))
    float n1 = Xout[o + 16] + mx1;
    Xout[o]       = n0;
    Xout[o + 16]  = n1;
    Xoutb[o]      = (__bf16)n0;      // bf16 shadow for next layer's WMMA A
    Xoutb[o + 16] = (__bf16)n1;
  }
}

// ---------------------------------------------------------------------------
// BN fold: scale = g*rsqrt(var+eps), shift = b - mu*scale
// ---------------------------------------------------------------------------
__global__ void dgcn_finalize_kernel(const float* __restrict__ sum,
                                     const float* __restrict__ sumsq,
                                     const float* __restrict__ gamma,
                                     const float* __restrict__ beta,
                                     float* __restrict__ scale,
                                     float* __restrict__ shift, float invR, int C) {
  int c = blockIdx.x * blockDim.x + threadIdx.x;
  if (c >= C) return;
  float mu  = sum[c] * invR;
  float var = sumsq[c] * invR - mu * mu;
  float s   = gamma[c] * rsqrtf(var + 1e-5f);
  scale[c] = s;
  shift[c] = beta[c] - mu * s;
}

// BN + LeakyReLU elementwise (branch-a path; aggregation added later).
__global__ void dgcn_bn_act_kernel(const float* __restrict__ H,
                                   const float* __restrict__ scale,
                                   const float* __restrict__ shift,
                                   float* __restrict__ Y, int Cout, int total) {
  int i = blockIdx.x * blockDim.x + threadIdx.x;
  if (i >= total) return;
  int c = i % Cout;
  Y[i] = lrelu(H[i] * scale[c] + shift[c]);
}

// ---------------------------------------------------------------------------
// Xout[g,c] += max_k lrelu(BN(Hb[nbr(g,k), c])); refreshes bf16 shadow.
// One thread = node x 4 channels (float4 gathers, L2-resident).
// ---------------------------------------------------------------------------
__global__ __launch_bounds__(256)
void dgcn_segmax_add_kernel(const float* __restrict__ Hb, const int* __restrict__ eidx,
                            const float* __restrict__ scale, const float* __restrict__ shift,
                            float* __restrict__ Xout, __bf16* __restrict__ Xoutb,
                            int N, int E, int Cout) {
  const int groups = Cout >> 2;
  const int tid = blockIdx.x * blockDim.x + threadIdx.x;
  const int g   = tid / groups;
  const int c4  = (tid - g * groups) << 2;
  const int b   = g / N;
  const int j   = g - b * N;
  const int* ep = eidx + E + (size_t)j * 16;
  const int base = b * N;
  const float4 sc = *(const float4*)(scale + c4);
  const float4 sh = *(const float4*)(shift + c4);
  float m0 = NEG_BIG, m1 = NEG_BIG, m2 = NEG_BIG, m3 = NEG_BIG;
#pragma unroll
  for (int k = 0; k < 16; ++k) {
    int nbr = ep[k] + base;
    const float4 h = *(const float4*)(Hb + (size_t)nbr * Cout + c4);
    m0 = fmaxf(m0, lrelu(h.x * sc.x + sh.x));
    m1 = fmaxf(m1, lrelu(h.y * sc.y + sh.y));
    m2 = fmaxf(m2, lrelu(h.z * sc.z + sh.z));
    m3 = fmaxf(m3, lrelu(h.w * sc.w + sh.w));
  }
  size_t o = (size_t)g * Cout + c4;
  float4* outp = (float4*)(Xout + o);
  float4 cur = *outp;
  cur.x += m0; cur.y += m1; cur.z += m2; cur.w += m3;
  *outp = cur;
  if (Xoutb) {
    Xoutb[o + 0] = (__bf16)cur.x;
    Xoutb[o + 1] = (__bf16)cur.y;
    Xoutb[o + 2] = (__bf16)cur.z;
    Xoutb[o + 3] = (__bf16)cur.w;
  }
}

// ---------------------------------------------------------------------------
// Head: one wave per row, lane-strided channels (coalesced), shfl reduce.
// ---------------------------------------------------------------------------
__global__ __launch_bounds__(256)
void dgcn_final_dot_kernel(const float* __restrict__ X0, const float* __restrict__ X1,
                           const float* __restrict__ X2, const float* __restrict__ X3,
                           const float* __restrict__ W4, float* __restrict__ H4,
                           float* __restrict__ sum, float* __restrict__ sumsq) {
  __shared__ float bs[2];
  if (threadIdx.x == 0) { bs[0] = 0.f; bs[1] = 0.f; }
  __syncthreads();
  const int lane = threadIdx.x & 31;
  const int wave = threadIdx.x >> 5;
  const int r = blockIdx.x * 8 + wave;
  float acc = 0.f;
  const float* p = X0 + (size_t)r * 64;
#pragma unroll
  for (int c = 0; c < 64; c += 32)  acc += p[c + lane] * W4[c + lane];
  p = X1 + (size_t)r * 128;
#pragma unroll
  for (int c = 0; c < 128; c += 32) acc += p[c + lane] * W4[64 + c + lane];
  p = X2 + (size_t)r * 256;
#pragma unroll
  for (int c = 0; c < 256; c += 32) acc += p[c + lane] * W4[192 + c + lane];
  p = X3 + (size_t)r * 256;
#pragma unroll
  for (int c = 0; c < 256; c += 32) acc += p[c + lane] * W4[448 + c + lane];
  for (int o = 16; o > 0; o >>= 1) acc += __shfl_down(acc, o, 32);
  if (lane == 0) {
    H4[r] = acc;
    atomicAdd(&bs[0], acc);
    atomicAdd(&bs[1], acc * acc);
  }
  __syncthreads();
  if (threadIdx.x == 0) { atomicAdd(&sum[0], bs[0]); atomicAdd(&sumsq[0], bs[1]); }
}

__global__ void dgcn_final_out_kernel(const float* __restrict__ H4,
                                      const float* __restrict__ scale,
                                      const float* __restrict__ shift,
                                      const float* __restrict__ bias,
                                      float* __restrict__ out, int total) {
  int r = blockIdx.x * blockDim.x + threadIdx.x;
  if (r >= total) return;
  out[r] = lrelu(H4[r] * scale[0] + shift[0]) + bias[0];
}

__global__ void dgcn_zero_kernel(float* __restrict__ p, int n) {
  int i = blockIdx.x * blockDim.x + threadIdx.x;
  if (i < n) p[i] = 0.f;
}

// ---------------------------------------------------------------------------
extern "C" void kernel_launch(void* const* d_in, const int* in_sizes, int n_in,
                              void* d_out, int out_size, void* d_ws, size_t ws_size,
                              hipStream_t stream) {
  (void)in_sizes; (void)n_in; (void)out_size; (void)ws_size;
  const int Nn = 16384, Bb = 2;
  const int Rn = Bb * Nn;        // 32768 node rows
  const int E  = Nn * 16;        // 262144 edges / batch
  const int RE = Bb * E;         // 524288 edge rows

  const float* x    = (const float*)d_in[0];
  const int*   eidx = (const int*)d_in[1];
  const float *W0a=(const float*)d_in[2],  *g0a=(const float*)d_in[3],  *b0a=(const float*)d_in[4];
  const float *W0b=(const float*)d_in[5],  *g0b=(const float*)d_in[6],  *b0b=(const float*)d_in[7];
  const float *W1a=(const float*)d_in[8],  *g1a=(const float*)d_in[9],  *b1a=(const float*)d_in[10];
  const float *W1b=(const float*)d_in[11], *g1b=(const float*)d_in[12], *b1b=(const float*)d_in[13];
  const float *W2a=(const float*)d_in[14], *g2a=(const float*)d_in[15], *b2a=(const float*)d_in[16];
  const float *W2b=(const float*)d_in[17], *g2b=(const float*)d_in[18], *b2b=(const float*)d_in[19];
  const float *W3a=(const float*)d_in[20], *g3a=(const float*)d_in[21], *b3a=(const float*)d_in[22];
  const float *W3b=(const float*)d_in[23], *g3b=(const float*)d_in[24], *b3b=(const float*)d_in[25];
  const float *W4 =(const float*)d_in[26], *g4 =(const float*)d_in[27], *b4 =(const float*)d_in[28];
  const float *bias=(const float*)d_in[29];
  float* out = (float*)d_out;

  float* ws = (float*)d_ws;
  size_t off = 0;
  auto alloc = [&](size_t nfloats) { float* p = ws + off; off += nfloats; return p; };
  auto allocb = [&](size_t nbf16)  { return (__bf16*)alloc(nbf16 / 2); };

  float* XR  = alloc((size_t)Rn * 32);
  float* X0  = alloc((size_t)Rn * 64);
  float* X1  = alloc((size_t)Rn * 128);
  float* X2  = alloc((size_t)Rn * 256);
  float* X3  = alloc((size_t)Rn * 256);
  float* HA  = alloc((size_t)Rn * 256);
  float* HB  = alloc((size_t)Rn * 256);
  float* H4  = alloc((size_t)Rn);
  __bf16* XRb = allocb((size_t)Rn * 32);    // bf16 shadows (WMMA A operands)
  __bf16* X0b = allocb((size_t)Rn * 64);
  __bf16* X1b = allocb((size_t)Rn * 128);
  __bf16* X2b = allocb((size_t)Rn * 256);
  __bf16* P0a = allocb(32 * 64);            // packed bf16 weights (B operands)
  __bf16* P0b = allocb(32 * 64);
  __bf16* P1a = allocb(64 * 128);
  __bf16* P1b = allocb(64 * 128);
  __bf16* P2a = allocb(128 * 256);
  __bf16* P2b = allocb(128 * 256);
  __bf16* P3a = allocb(256 * 256);
  __bf16* P3b = allocb(256 * 256);
  float* SUMA = alloc(256); float* SQA = alloc(256);   // SUMA..SQB contiguous (1024)
  float* SUMB = alloc(256); float* SQB = alloc(256);
  float* SCA  = alloc(256); float* SHA = alloc(256);
  float* SCB  = alloc(256); float* SHB = alloc(256);

  auto packw = [&](const float* W, __bf16* WP, int Cin, int Cout) {
    dgcn_pack_w_kernel<<<(Cin * Cout + 255) / 256, 256, 0, stream>>>(W, WP, Cin, Cout);
  };
  packw(W0a, P0a, 32, 64);   packw(W0b, P0b, 32, 64);
  packw(W1a, P1a, 64, 128);  packw(W1b, P1b, 64, 128);
  packw(W2a, P2a, 128, 256); packw(W2b, P2b, 128, 256);
  packw(W3a, P3a, 256, 256); packw(W3b, P3b, 256, 256);

  dgcn_transpose_kernel<<<dim3(Nn / 32, Bb), dim3(32, 32), 0, stream>>>(x, XR, XRb, Nn);

  // ----- layer 0 -----
  dgcn_zero_kernel<<<4, 256, 0, stream>>>(SUMA, 1024);
  dgcn_gemm_stats_kernel<<<dim3(Rn / 256, 64 / 32), 256, 0, stream>>>(XRb, P0a, HA, SUMA, SQA, 32, 64);
  dgcn_edge_stats_kernel<<<dim3(Rn / 8, 64 / 32), 256, 0, stream>>>(XR, P0b, eidx, SUMB, SQB, Nn, E, 64);
  dgcn_finalize_kernel<<<1, 256, 0, stream>>>(SUMA, SQA, g0a, b0a, SCA, SHA, 1.f / Rn, 64);
  dgcn_finalize_kernel<<<1, 256, 0, stream>>>(SUMB, SQB, g0b, b0b, SCB, SHB, 1.f / RE, 64);
  dgcn_bn_act_kernel<<<(Rn * 64 + 255) / 256, 256, 0, stream>>>(HA, SCA, SHA, X0, 64, Rn * 64);
  dgcn_edge_apply_kernel<<<dim3(Rn / 8, 64 / 32), 256, 0, stream>>>(XR, P0b, eidx, SCB, SHB, X0, X0b, Nn, E, 64);

  // ----- layers 1..3 -----
  auto layer = [&](const __bf16* Xinb, int Cin, float* Xout, __bf16* Xoutb, int Cout,
                   const __bf16* Pa, const float* ga, const float* ba,
                   const __bf16* Pb, const float* gb, const float* bb) {
    dgcn_zero_kernel<<<4, 256, 0, stream>>>(SUMA, 1024);
    dgcn_gemm_stats_kernel<<<dim3(Rn / 256, Cout / 32), 256, 0, stream>>>(Xinb, Pa, HA, SUMA, SQA, Cin, Cout);
    dgcn_gemm_stats_kernel<<<dim3(Rn / 256, Cout / 32), 256, 0, stream>>>(Xinb, Pb, HB, SUMB, SQB, Cin, Cout);
    dgcn_finalize_kernel<<<1, 256, 0, stream>>>(SUMA, SQA, ga, ba, SCA, SHA, 1.f / Rn, Cout);
    dgcn_finalize_kernel<<<1, 256, 0, stream>>>(SUMB, SQB, gb, bb, SCB, SHB, 1.f / Rn, Cout);
    dgcn_bn_act_kernel<<<(Rn * Cout + 255) / 256, 256, 0, stream>>>(HA, SCA, SHA, Xout, Cout, Rn * Cout);
    dgcn_segmax_add_kernel<<<(Rn * (Cout / 4) + 255) / 256, 256, 0, stream>>>(HB, eidx, SCB, SHB, Xout, Xoutb, Nn, E, Cout);
  };
  layer(X0b, 64,  X1, X1b, 128, P1a, g1a, b1a, P1b, g1b, b1b);
  layer(X1b, 128, X2, X2b, 256, P2a, g2a, b2a, P2b, g2b, b2b);
  layer(X2b, 256, X3, (__bf16*)nullptr, 256, P3a, g3a, b3a, P3b, g3b, b3b);

  // ----- head -----
  dgcn_zero_kernel<<<4, 256, 0, stream>>>(SUMA, 1024);
  dgcn_final_dot_kernel<<<Rn / 8, 256, 0, stream>>>(X0, X1, X2, X3, W4, H4, SUMA, SQA);
  dgcn_finalize_kernel<<<1, 32, 0, stream>>>(SUMA, SQA, g4, b4, SCA, SHA, 1.f / Rn, 1);
  dgcn_final_out_kernel<<<Rn / 256, 256, 0, stream>>>(H4, SCA, SHA, bias, out, Rn);
}
